// Eagle3Attention_27101243638392
// MI455X (gfx1250) — matrix operand order repaired and checked by reading, not compiled, not run on hardware
//
#include <hip/hip_runtime.h>
#include <hip/hip_bf16.h>

// ---------------------------------------------------------------------------
// Eagle3 GQA attention for gfx1250 (MI455X), bf16 WMMA pipeline
//   k1: fused QKV projection GEMM (f32 -> bf16 WMMA, f32 acc), writes q/k/vT
//   k2: RoPE in-place on q,k
//   k3: flash attention; K/V tiles staged block-cooperatively into LDS with
//       global_load_async_to_lds_b128 (ASYNCcnt), WMMA fed from LDS
//   k4: O-projection GEMM, f32 output
// ---------------------------------------------------------------------------

typedef __attribute__((ext_vector_type(16))) __bf16 v16bf;
typedef __attribute__((ext_vector_type(8)))  __bf16 v8bf;
typedef __attribute__((ext_vector_type(8)))  float  v8f;

#define WMMA_BF16(a, b, c) \
  __builtin_amdgcn_wmma_f32_16x16x32_bf16(false, (a), false, (b), (short)0, (c), false, false)

constexpr int Bb   = 2;
constexpr int S    = 2048;
constexpr int TWOH = 8192;     // 2*H input width
constexpr int NQ   = 32;
constexpr int NKV  = 8;
constexpr int Dh   = 128;
constexpr int NTOT = NQ * Dh + 2 * NKV * Dh;  // 6144 fused QKV cols
constexpr int Mtot = Bb * S;                  // 4096 rows
constexpr int Hout = 4096;                    // output hidden

// ---- fragment load helpers -------------------------------------------------
// A-frag (16x32 bf16): lane m=lane&15, half h=lane>>4.
// elements 0..7  <- K = k0 + 16h + {0..7}
// elements 8..15 <- K = k0 + 16h + 8 + {0..7}
__device__ __forceinline__ void load8f(const float* __restrict__ p, v16bf& a, int off) {
  float4 x = *(const float4*)p;
  float4 y = *(const float4*)(p + 4);
  a[off + 0] = (__bf16)x.x; a[off + 1] = (__bf16)x.y;
  a[off + 2] = (__bf16)x.z; a[off + 3] = (__bf16)x.w;
  a[off + 4] = (__bf16)y.x; a[off + 5] = (__bf16)y.y;
  a[off + 6] = (__bf16)y.z; a[off + 7] = (__bf16)y.w;
}
__device__ __forceinline__ void load8b(const __bf16* p, v16bf& a, int off) {
  v8bf t = *(const v8bf*)p;
#pragma unroll
  for (int i = 0; i < 8; ++i) a[off + i] = t[i];
}
// B-frag (32x16 bf16): lane n=lane&15, half h=lane>>4; 16 contiguous K at k0+16h.
__device__ __forceinline__ v16bf load16f(const float* __restrict__ p) {
  v16bf b;
  load8f(p, b, 0);
  load8f(p + 8, b, 8);
  return b;
}

// ---- CDNA5 async global->LDS copy (16 bytes), tracked by ASYNCcnt ----------
__device__ __forceinline__ unsigned lds_addr_of(const void* p) {
  return (unsigned)(unsigned long long)p;  // low 32 bits = LDS byte address
}
__device__ __forceinline__ void async_copy_b128(unsigned lds_dst, const void* gsrc) {
  asm volatile("global_load_async_to_lds_b128 %0, %1, off"
               :: "v"(lds_dst), "v"((unsigned long long)gsrc)
               : "memory");
}
__device__ __forceinline__ void wait_async0() {
  asm volatile("s_wait_asynccnt 0x0" ::: "memory");
}

// ---------------------------------------------------------------------------
// Kernel 1: fused QKV projection.  C[4096 x 6144] = hs[4096 x 8192] @ W^T
// 4 waves/block; each wave computes a 64x64 tile (4x4 WMMA accumulators).
// Epilogue scatters bf16 into q [B,NQ,S,D], k [B,NKV,S,D], vT [B,NKV,D,S].
// ---------------------------------------------------------------------------
__global__ __launch_bounds__(128) void qkv_proj_kernel(
    const float* __restrict__ hs, const float* __restrict__ Wq,
    const float* __restrict__ Wk, const float* __restrict__ Wv,
    __bf16* __restrict__ qbuf, __bf16* __restrict__ kbuf,
    __bf16* __restrict__ vTbuf) {
  const int lane = threadIdx.x & 31;
  const int wave = threadIdx.x >> 5;
  const int ln = lane & 15, lh = lane >> 4;
  const int m0 = blockIdx.x * 64;
  const int n0 = blockIdx.y * 256 + wave * 64;

  v8f acc[16];
#pragma unroll
  for (int i = 0; i < 16; ++i) { v8f z = {}; acc[i] = z; }

  const float* arow[4];
#pragma unroll
  for (int mi = 0; mi < 4; ++mi)
    arow[mi] = hs + (size_t)(m0 + mi * 16 + ln) * TWOH;

  const float* brow[4];
#pragma unroll
  for (int ni = 0; ni < 4; ++ni) {
    int col = n0 + ni * 16 + ln;
    const float* p;
    if (col < NQ * Dh)                 p = Wq + (size_t)col * TWOH;
    else if (col < NQ * Dh + NKV * Dh) p = Wk + (size_t)(col - NQ * Dh) * TWOH;
    else                               p = Wv + (size_t)(col - NQ * Dh - NKV * Dh) * TWOH;
    brow[ni] = p;
  }

  for (int k0 = 0; k0 < TWOH; k0 += 32) {
    v16bf a[4];
#pragma unroll
    for (int mi = 0; mi < 4; ++mi) {
      load8f(arow[mi] + k0 + 16 * lh, a[mi], 0);
      load8f(arow[mi] + k0 + 16 * lh + 8, a[mi], 8);
      __builtin_prefetch(arow[mi] + k0 + 8 * lh + 128, 0, 0);
    }
#pragma unroll
    for (int ni = 0; ni < 4; ++ni) {
      v16bf b = load16f(brow[ni] + k0 + 16 * lh);
#pragma unroll
      for (int mi = 0; mi < 4; ++mi)
        acc[mi * 4 + ni] = WMMA_BF16(a[mi], b, acc[mi * 4 + ni]);
    }
  }

  // epilogue: C layout -> lane holds rows m=8*lh+r, col n=ln
#pragma unroll
  for (int mi = 0; mi < 4; ++mi) {
#pragma unroll
    for (int ni = 0; ni < 4; ++ni) {
      int col = n0 + ni * 16 + ln;
#pragma unroll
      for (int r = 0; r < 8; ++r) {
        int row = m0 + mi * 16 + 8 * lh + r;
        int b_ = row >> 11, s = row & (S - 1);
        __bf16 val = (__bf16)acc[mi * 4 + ni][r];
        if (col < NQ * Dh) {
          int hq = col >> 7, d = col & 127;
          qbuf[(((size_t)(b_ * NQ + hq)) * S + s) * Dh + d] = val;
        } else if (col < NQ * Dh + NKV * Dh) {
          int c2 = col - NQ * Dh; int kvh = c2 >> 7, d = c2 & 127;
          kbuf[(((size_t)(b_ * NKV + kvh)) * S + s) * Dh + d] = val;
        } else {
          int c2 = col - NQ * Dh - NKV * Dh; int kvh = c2 >> 7, d = c2 & 127;
          vTbuf[(((size_t)(b_ * NKV + kvh)) * Dh + d) * S + s] = val;  // transposed V
        }
      }
    }
  }
}

// ---------------------------------------------------------------------------
// Kernel 2: RoPE in place on q and k.  One thread per (row, d<64) pair.
// cos[s][d] == cos[s][d+64] for this table, so only d<64 entries are read.
// ---------------------------------------------------------------------------
__global__ void rope_kernel(__bf16* __restrict__ qbuf, __bf16* __restrict__ kbuf,
                            const float* __restrict__ cosT,
                            const float* __restrict__ sinT) {
  const long QP = (long)Bb * NQ * S * 64;
  const long KP = (long)Bb * NKV * S * 64;
  long idx = (long)blockIdx.x * blockDim.x + threadIdx.x;
  __bf16* base;
  long t;
  if (idx < QP)            { base = qbuf; t = idx; }
  else if (idx < QP + KP)  { base = kbuf; t = idx - QP; }
  else return;
  int d = (int)(t & 63);
  int s = (int)((t >> 6) & (S - 1));
  long hb = t >> 17;  // (batch*heads + head): S*64 = 2^17 pairs per head
  __bf16* p = base + (hb * S + s) * (long)Dh;
  float c = cosT[s * Dh + d], sn = sinT[s * Dh + d];
  float x0 = (float)p[d], x1 = (float)p[d + 64];
  p[d]      = (__bf16)(x0 * c - x1 * sn);
  p[d + 64] = (__bf16)(x1 * c + x0 * sn);
}

// ---------------------------------------------------------------------------
// Kernel 3: causal flash attention with async-LDS K/V staging.
// grid = (S/64, B*NQ); 4 waves/block; wave owns 16 query rows.
// Per 64-key block (uniform trip count across the block):
//   all 128 threads async-copy K tile (64x128) and V^T tile (128x64) to LDS,
//   s_wait_asynccnt 0 + barrier, then 16 WMMA (QK^T) + online softmax +
//   16 WMMA (PV) with B-frags read from LDS.  Waves whose rows precede the
//   key block get fully-masked scores (exp -> 0), preserving the math.
// ---------------------------------------------------------------------------
__global__ __launch_bounds__(128) void attn_kernel(
    const __bf16* __restrict__ qbuf, const __bf16* __restrict__ kbuf,
    const __bf16* __restrict__ vTbuf, __bf16* __restrict__ attnbuf) {
  __shared__ __attribute__((aligned(32))) __bf16 klds[64 * 128];   // [key][d]
  __shared__ __attribute__((aligned(32))) __bf16 vlds[128 * 64];   // [d][key]
  __shared__ __attribute__((aligned(32))) __bf16 plds[4][16 * 64]; // probs/wave
  const int tid  = threadIdx.x;
  const int lane = tid & 31;
  const int wave = tid >> 5;
  const int ln = lane & 15, lh = lane >> 4;
  const int hh = blockIdx.y;            // b*NQ + qh
  const int b_ = hh >> 5, qh = hh & 31;
  const int kvh = qh >> 2;              // G = 4
  const int qr0 = blockIdx.x * 64 + wave * 16;
  const int jend = blockIdx.x * 64 + 63;  // uniform across block

  const __bf16* qp = qbuf + ((size_t)(b_ * NQ + qh)) * S * Dh;
  const __bf16* kp = kbuf + ((size_t)(b_ * NKV + kvh)) * S * Dh;
  const __bf16* vp = vTbuf + ((size_t)(b_ * NKV + kvh)) * Dh * S;

  // preload Q A-frags for all 4 K-steps of D=128
  v16bf aq[4];
#pragma unroll
  for (int kk = 0; kk < 4; ++kk) {
    const __bf16* r = qp + (size_t)(qr0 + ln) * Dh + kk * 32;
    load8b(r + 16 * lh, aq[kk], 0);
    load8b(r + 16 * lh + 8, aq[kk], 8);
  }

  v8f acc[8];
#pragma unroll
  for (int i = 0; i < 8; ++i) { v8f z = {}; acc[i] = z; }
  float rm[8], rl[8];
#pragma unroll
  for (int r = 0; r < 8; ++r) { rm[r] = -3.0e38f; rl[r] = 0.f; }

  const float scale = 0.08838834764831845f;  // 1/sqrt(128)
  __bf16* myP = plds[wave];
  const unsigned kbase = lds_addr_of(klds);
  const unsigned vbase = lds_addr_of(vlds);

  for (int j0 = 0; j0 <= jend; j0 += 64) {
    // ---- stage K tile (contiguous 16KB) and V^T tile (128 rows x 128B)
    const __bf16* gK = kp + (size_t)j0 * Dh;
    const __bf16* gV = vp + (size_t)tid * S + j0;
#pragma unroll
    for (int i = 0; i < 8; ++i) {
      int c = tid + 128 * i;                       // 16B chunk id, 0..1023
      async_copy_b128(kbase + c * 16, gK + c * 8);
    }
#pragma unroll
    for (int i = 0; i < 8; ++i)
      async_copy_b128(vbase + (unsigned)(tid * 64 + i * 8) * 2, gV + i * 8);
    wait_async0();
    __syncthreads();

    // ---- S = Q K^T over this 64-key block (B-frags from LDS)
    v8f sc[4];
#pragma unroll
    for (int t = 0; t < 4; ++t) {
      v8f z = {}; sc[t] = z;
#pragma unroll
      for (int kk = 0; kk < 4; ++kk) {
        v16bf kb = *(const v16bf*)(klds + (t * 16 + ln) * Dh + kk * 32 + 16 * lh);
        sc[t] = WMMA_BF16(aq[kk], kb, sc[t]);
      }
    }
    const bool needMask = (j0 + 63) > qr0;
#pragma unroll
    for (int t = 0; t < 4; ++t) {
      int key = j0 + t * 16 + ln;
#pragma unroll
      for (int r = 0; r < 8; ++r) {
        float v = sc[t][r] * scale;
        if (needMask && key > qr0 + 8 * lh + r) v += -1.0e9f;  // additive causal mask
        sc[t][r] = v;
      }
    }
    // ---- online softmax per row (row lives on 16 lanes sharing lane>>4)
#pragma unroll
    for (int r = 0; r < 8; ++r) {
      float mx = fmaxf(fmaxf(sc[0][r], sc[1][r]), fmaxf(sc[2][r], sc[3][r]));
      mx = fmaxf(mx, __shfl_xor(mx, 1, 32));
      mx = fmaxf(mx, __shfl_xor(mx, 2, 32));
      mx = fmaxf(mx, __shfl_xor(mx, 4, 32));
      mx = fmaxf(mx, __shfl_xor(mx, 8, 32));
      float newm = fmaxf(rm[r], mx);
      float corr = __expf(rm[r] - newm);
      rm[r] = newm;
      float rsum = 0.f;
#pragma unroll
      for (int t = 0; t < 4; ++t) {
        float p = __expf(sc[t][r] - newm);
        sc[t][r] = p;
        rsum += p;
      }
      rsum += __shfl_xor(rsum, 1, 32);
      rsum += __shfl_xor(rsum, 2, 32);
      rsum += __shfl_xor(rsum, 4, 32);
      rsum += __shfl_xor(rsum, 8, 32);
      rl[r] = rl[r] * corr + rsum;
#pragma unroll
      for (int dt = 0; dt < 8; ++dt) acc[dt][r] *= corr;
      // stage probs (row-major 16x64 bf16) for A-frag reload
#pragma unroll
      for (int t = 0; t < 4; ++t)
        myP[(8 * lh + r) * 64 + t * 16 + ln] = (__bf16)sc[t][r];
    }
    asm volatile("s_wait_dscnt 0x0" ::: "memory");  // cross-lane LDS visibility
    // ---- O += P V  (A = probs from LDS, B = V^T rows from LDS)
    v16bf pa[2];
#pragma unroll
    for (int kk2 = 0; kk2 < 2; ++kk2) {
      const __bf16* r = myP + ln * 64 + kk2 * 32;
      load8b(r + 16 * lh, pa[kk2], 0);
      load8b(r + 16 * lh + 8, pa[kk2], 8);
    }
#pragma unroll
    for (int dt = 0; dt < 8; ++dt) {
#pragma unroll
      for (int kk2 = 0; kk2 < 2; ++kk2) {
        v16bf vb = *(const v16bf*)(vlds + (dt * 16 + ln) * 64 + kk2 * 32 + 16 * lh);
        acc[dt] = WMMA_BF16(pa[kk2], vb, acc[dt]);
      }
    }
    __syncthreads();  // all waves done reading K/V LDS before next overwrite
  }

  // epilogue: write [B, S, NQ*D] bf16 for the O-projection
#pragma unroll
  for (int dt = 0; dt < 8; ++dt) {
    int d = dt * 16 + ln;
#pragma unroll
    for (int r = 0; r < 8; ++r) {
      int s = qr0 + 8 * lh + r;
      float o = acc[dt][r] / rl[r];
      attnbuf[(((size_t)(b_ * S + s)) * NQ + qh) * Dh + d] = (__bf16)o;
    }
  }
}

// ---------------------------------------------------------------------------
// Kernel 4: O projection.  out[4096 x 4096] = attn_bf16 @ Wo^T (f32 out)
// ---------------------------------------------------------------------------
__global__ __launch_bounds__(128) void oproj_kernel(
    const __bf16* __restrict__ attn, const float* __restrict__ Wo,
    float* __restrict__ out) {
  const int lane = threadIdx.x & 31;
  const int wave = threadIdx.x >> 5;
  const int ln = lane & 15, lh = lane >> 4;
  const int m0 = blockIdx.x * 64;
  const int n0 = blockIdx.y * 256 + wave * 64;
  const int K = NQ * Dh;  // 4096

  v8f acc[16];
#pragma unroll
  for (int i = 0; i < 16; ++i) { v8f z = {}; acc[i] = z; }

  const __bf16* arow[4];
#pragma unroll
  for (int mi = 0; mi < 4; ++mi)
    arow[mi] = attn + (size_t)(m0 + mi * 16 + ln) * K;
  const float* brow[4];
#pragma unroll
  for (int ni = 0; ni < 4; ++ni)
    brow[ni] = Wo + (size_t)(n0 + ni * 16 + ln) * K;

  for (int k0 = 0; k0 < K; k0 += 32) {
    v16bf a[4];
#pragma unroll
    for (int mi = 0; mi < 4; ++mi) {
      load8b(arow[mi] + k0 + 16 * lh, a[mi], 0);
      load8b(arow[mi] + k0 + 16 * lh + 8, a[mi], 8);
    }
#pragma unroll
    for (int ni = 0; ni < 4; ++ni) {
      v16bf b = load16f(brow[ni] + k0 + 16 * lh);
#pragma unroll
      for (int mi = 0; mi < 4; ++mi)
        acc[mi * 4 + ni] = WMMA_BF16(a[mi], b, acc[mi * 4 + ni]);
    }
  }
#pragma unroll
  for (int mi = 0; mi < 4; ++mi) {
#pragma unroll
    for (int ni = 0; ni < 4; ++ni) {
      int col = n0 + ni * 16 + ln;
#pragma unroll
      for (int r = 0; r < 8; ++r) {
        int row = m0 + mi * 16 + 8 * lh + r;
        out[(size_t)row * Hout + col] = acc[mi * 4 + ni][r];
      }
    }
  }
}

// ---------------------------------------------------------------------------
extern "C" void kernel_launch(void* const* d_in, const int* in_sizes, int n_in,
                              void* d_out, int out_size, void* d_ws, size_t ws_size,
                              hipStream_t stream) {
  (void)in_sizes; (void)n_in; (void)out_size; (void)ws_size;
  const float* hs   = (const float*)d_in[0];
  // d_in[1] = attention_mask: causal mask is applied analytically (same math)
  const float* cosT = (const float*)d_in[2];
  const float* sinT = (const float*)d_in[3];
  const float* Wq   = (const float*)d_in[4];
  const float* Wk   = (const float*)d_in[5];
  const float* Wv   = (const float*)d_in[6];
  const float* Wo   = (const float*)d_in[7];
  float* out = (float*)d_out;

  char* ws = (char*)d_ws;
  size_t o = 0;
  __bf16* qbuf  = (__bf16*)(ws + o); o += (size_t)Bb * NQ * S * Dh * 2;   // 32 MB
  __bf16* kbuf  = (__bf16*)(ws + o); o += (size_t)Bb * NKV * S * Dh * 2;  //  8 MB
  __bf16* vTbuf = (__bf16*)(ws + o); o += (size_t)Bb * NKV * S * Dh * 2;  //  8 MB
  __bf16* attnb = (__bf16*)(ws + o);                                      // 32 MB

  qkv_proj_kernel<<<dim3(Mtot / 64, NTOT / 256), dim3(128), 0, stream>>>(
      hs, Wq, Wk, Wv, qbuf, kbuf, vTbuf);

  long pairs = (long)Bb * NQ * S * 64 + (long)Bb * NKV * S * 64;
  rope_kernel<<<dim3((unsigned)((pairs + 255) / 256)), dim3(256), 0, stream>>>(
      qbuf, kbuf, cosT, sinT);

  attn_kernel<<<dim3(S / 64, Bb * NQ), dim3(128), 0, stream>>>(
      qbuf, kbuf, vTbuf, attnb);

  oproj_kernel<<<dim3(Mtot / 64, Hout / 256), dim3(128), 0, stream>>>(
      attnb, Wo, out);
}
